// Layer_VanillaMHA_48155173323384
// MI455X (gfx1250) — matrix operand order repaired and checked by reading, not compiled, not run on hardware
//
#include <hip/hip_runtime.h>
#include <math.h>

// ---------------- problem constants ----------------
#define NN    4096   // nodes
#define IND   512    // input dim
#define KD    64     // key dim per head
#define NH    8      // heads
#define ODIM  512    // output dim

typedef __attribute__((ext_vector_type(16))) __bf16 bf16x16;
typedef __attribute__((ext_vector_type(8)))  __bf16 bf16x8;
typedef __attribute__((ext_vector_type(8)))  float  f32x8;

static __device__ __forceinline__ bf16x16 cat16(bf16x8 lo, bf16x8 hi) {
  union { bf16x8 h[2]; bf16x16 v; } u;
  u.h[0] = lo; u.h[1] = hi;
  return u.v;
}
// B operand: 16 contiguous bf16 starting at p (lane-private address)
static __device__ __forceinline__ bf16x16 ldB16(const __bf16* p) {
  return cat16(*(const bf16x8*)p, *(const bf16x8*)(p + 8));
}
// A operand: 16 contiguous K (K=16h..16h+15)
static __device__ __forceinline__ bf16x16 ldA16(const __bf16* chunk_base, int hf) {
  return cat16(*(const bf16x8*)(chunk_base + 16 * hf),
               *(const bf16x8*)(chunk_base + 16 * hf + 8));
}
static __device__ __forceinline__ f32x8 wmma_bf16(bf16x16 a, bf16x16 b, f32x8 c) {
  return __builtin_amdgcn_wmma_f32_16x16x32_bf16(false, a, false, b, (short)0, c,
                                                 false, false);
}
static __device__ __forceinline__ f32x8 zero8() {
  f32x8 z = {0.f, 0.f, 0.f, 0.f, 0.f, 0.f, 0.f, 0.f};
  return z;
}

// ---------------- kernel 0a: f32 -> bf16 elementwise ----------------
__global__ void __launch_bounds__(256)
k_cvt_bf16(const float* __restrict__ in, __bf16* __restrict__ out, int n) {
  int i = blockIdx.x * blockDim.x + threadIdx.x;
  if (i < n) out[i] = (__bf16)in[i];
}

// ---------------- kernel 0b: transpose f32 [B][R][C] -> bf16 [B][C][R] ----
__global__ void __launch_bounds__(256)
k_transpose_bf16(const float* __restrict__ in, __bf16* __restrict__ out,
                 int Bn, int R, int C) {
  int i = blockIdx.x * blockDim.x + threadIdx.x;
  int total = Bn * R * C;
  if (i >= total) return;
  int b = i / (R * C);
  int rem = i - b * (R * C);
  int c = rem / R;
  int r = rem - c * R;
  out[i] = (__bf16)in[((size_t)b * R + r) * C + c];
}

// ---------------- kernel 1: Q/K/V projections ----------------
// grid = (256 row-tiles, 8 heads, 3 matrices), block = 32 (one wave)
// Q,K row-major bf16 [h][4096][64]; V transposed bf16 [h][64][4096]
__global__ void __launch_bounds__(32)
k_proj(const __bf16* __restrict__ node,
       const __bf16* __restrict__ WqT, const __bf16* __restrict__ WkT,
       const __bf16* __restrict__ WvT,
       __bf16* __restrict__ Q, __bf16* __restrict__ K, __bf16* __restrict__ VT) {
  const int lane = threadIdx.x & 31;
  const int m16 = lane & 15;
  const int hf = lane >> 4;
  const int row0 = blockIdx.x * 16;
  const int head = blockIdx.y;
  const int mat = blockIdx.z;
  const __bf16* W =
      (mat == 0 ? WqT : (mat == 1 ? WkT : WvT)) + (size_t)head * KD * IND;

  f32x8 acc[4];
#pragma unroll
  for (int nt = 0; nt < 4; ++nt) acc[nt] = zero8();

#pragma unroll 4
  for (int c = 0; c < 16; ++c) {
    const __bf16* ar = node + (size_t)(row0 + m16) * IND + 32 * c;
    bf16x16 A = ldA16(ar, hf);
#pragma unroll
    for (int nt = 0; nt < 4; ++nt) {
      const __bf16* br = W + (size_t)(16 * nt + m16) * IND + 32 * c + 16 * hf;
      acc[nt] = wmma_bf16(A, ldB16(br), acc[nt]);
    }
  }

  if (mat == 2) {
    // V transposed: lane writes 8 consecutive bf16 (rows m = 8*hf..8*hf+7)
#pragma unroll
    for (int nt = 0; nt < 4; ++nt) {
      bf16x8 pk;
#pragma unroll
      for (int r = 0; r < 8; ++r) pk[r] = (__bf16)acc[nt][r];
      *(bf16x8*)(VT + ((size_t)head * KD + 16 * nt + m16) * NN + row0 + 8 * hf) = pk;
    }
  } else {
    const float sc = (mat == 0) ? 0.125f : 1.0f;  // kd^-0.5 folded into Q
    __bf16* O = (mat == 0 ? Q : K) + ((size_t)head * NN + row0) * KD;
#pragma unroll
    for (int nt = 0; nt < 4; ++nt)
#pragma unroll
      for (int r = 0; r < 8; ++r)
        O[(size_t)(r + 8 * hf) * KD + 16 * nt + m16] = (__bf16)(acc[nt][r] * sc);
  }
}

// ---------------- kernel 2: fused masked flash attention ----------------
// grid = 256 query blocks of 16; block = 256 = 8 waves, wave w == head w.
// Adjacency 16x32 tile staged once in LDS, shared by all 8 heads.
__global__ void __launch_bounds__(256)
k_flash(const __bf16* __restrict__ Q, const __bf16* __restrict__ K,
        const __bf16* __restrict__ VT, const float* __restrict__ adj,
        __bf16* __restrict__ HV) {
  __shared__ float adjt[16 * 32];
  __shared__ __bf16 pls[NH * 16 * 40];  // per-wave P tile, row stride 40 bf16

  const int tid = threadIdx.x;
  const int head = tid >> 5;
  const int lane = tid & 31;
  const int m16 = lane & 15;
  const int hf = lane >> 4;
  const int qb = blockIdx.x * 16;

  const __bf16* Qh = Q + ((size_t)head * NN + qb) * KD;
  const __bf16* Kh = K + (size_t)head * NN * KD;
  const __bf16* VTh = VT + (size_t)head * KD * NN;

  // Q block as two A operands (kd chunks 0..31, 32..63), resident all loop
  bf16x16 qa[2];
#pragma unroll
  for (int c = 0; c < 2; ++c)
    qa[c] = ldA16(Qh + (size_t)m16 * KD + 32 * c, hf);

  float mrun[8], lrun[8];
#pragma unroll
  for (int r = 0; r < 8; ++r) { mrun[r] = -__builtin_inff(); lrun[r] = 0.0f; }
  f32x8 oacc[4];
#pragma unroll
  for (int nt = 0; nt < 4; ++nt) oacc[nt] = zero8();

  __bf16* pw = &pls[head * 16 * 40];

  for (int kb = 0; kb < NN; kb += 32) {
    __syncthreads();  // previous adjt consumers done
    {
      const int row = tid >> 4;            // 0..15
      const int col = (tid & 15) * 2;      // 0..30 even
      const float2 a2 =
          *(const float2*)(adj + (size_t)(qb + row) * NN + kb + col);
      adjt[row * 32 + col] = a2.x;
      adjt[row * 32 + col + 1] = a2.y;
      if (kb + 32 < NN)
        __builtin_prefetch(adj + (size_t)(qb + row) * NN + kb + 32 + col, 0, 1);
    }
    __syncthreads();

    // ---- S = (Q*kd^-0.5) @ K^T : 16x32 chunk, two 16x16 N-tiles ----
    f32x8 s[2];
#pragma unroll
    for (int t = 0; t < 2; ++t) {
      f32x8 a = zero8();
#pragma unroll
      for (int c = 0; c < 2; ++c) {
        const __bf16* kr =
            Kh + (size_t)(kb + 16 * t + m16) * KD + 32 * c + 16 * hf;
        a = wmma_bf16(qa[c], ldB16(kr), a);
      }
      s[t] = a;
    }

    // ---- fused identity bias + adjacency mask (pre-softmax) ----
#pragma unroll
    for (int t = 0; t < 2; ++t)
#pragma unroll
      for (int r = 0; r < 8; ++r) {
        const int qg = qb + r + 8 * hf;
        const int kg = kb + 16 * t + m16;
        float a = adjt[(r + 8 * hf) * 32 + 16 * t + m16];
        const float e = (qg == kg) ? 1.0f : 0.0f;
        a = fmaxf(a, e);  // self-loops always unmasked
        s[t][r] = s[t][r] + 0.01f * e - 1e16f * (1.0f - a);
      }

    // ---- online softmax: rows live in half-waves of 16 lanes ----
    float alpha[8];
#pragma unroll
    for (int r = 0; r < 8; ++r) {
      float mx = fmaxf(s[0][r], s[1][r]);
#pragma unroll
      for (int off = 1; off < 16; off <<= 1)
        mx = fmaxf(mx, __shfl_xor(mx, off, 32));
      const float mnew = fmaxf(mrun[r], mx);
      alpha[r] = __expf(mrun[r] - mnew);
      mrun[r] = mnew;
      const float p0 = __expf(s[0][r] - mnew);
      const float p1 = __expf(s[1][r] - mnew);
      s[0][r] = p0;
      s[1][r] = p1;
      float rs = p0 + p1;
#pragma unroll
      for (int off = 1; off < 16; off <<= 1) rs += __shfl_xor(rs, off, 32);
      lrun[r] = lrun[r] * alpha[r] + rs;
    }
#pragma unroll
    for (int nt = 0; nt < 4; ++nt)
#pragma unroll
      for (int r = 0; r < 8; ++r) oacc[nt][r] *= alpha[r];

    // ---- D-layout -> A-layout for P via per-wave LDS tile ----
#pragma unroll
    for (int t = 0; t < 2; ++t)
#pragma unroll
      for (int r = 0; r < 8; ++r)
        pw[(size_t)(r + 8 * hf) * 40 + 16 * t + m16] = (__bf16)s[t][r];

    // cross-lane within one wave: DS ops are in-order, just drain DScnt
    asm volatile("s_wait_dscnt 0" ::: "memory");

    bf16x16 pa = ldA16(pw + (size_t)m16 * 40, hf);

    // ---- O += P @ V (4 N-tiles over kd=64) ----
#pragma unroll
    for (int nt = 0; nt < 4; ++nt) {
      const __bf16* vr = VTh + (size_t)(16 * nt + m16) * NN + kb + 16 * hf;
      oacc[nt] = wmma_bf16(pa, ldB16(vr), oacc[nt]);
    }
  }

  // ---- finalize: O / l, store per-head context bf16 [h][4096][64] ----
  float invl[8];
#pragma unroll
  for (int r = 0; r < 8; ++r) invl[r] = 1.0f / lrun[r];
  __bf16* HVh = HV + ((size_t)head * NN + qb) * KD;
#pragma unroll
  for (int nt = 0; nt < 4; ++nt)
#pragma unroll
    for (int r = 0; r < 8; ++r)
      HVh[(size_t)(r + 8 * hf) * KD + 16 * nt + m16] =
          (__bf16)(oacc[nt][r] * invl[r]);
}

// ---------------- kernel 3: output projection + head-sum ----------------
// grid = (256 row-tiles, 32 col-tiles), block = 32. K = 8 heads * 64 = 512.
__global__ void __launch_bounds__(32)
k_outproj(const __bf16* __restrict__ HV, const __bf16* __restrict__ WoT,
          float* __restrict__ out) {
  const int lane = threadIdx.x & 31;
  const int m16 = lane & 15;
  const int hf = lane >> 4;
  const int row0 = blockIdx.x * 16;
  const int col0 = blockIdx.y * 16;

  f32x8 acc = zero8();
#pragma unroll
  for (int head = 0; head < NH; ++head) {
    const __bf16* A0 = HV + ((size_t)head * NN + row0 + m16) * KD;
    const __bf16* B0 = WoT + ((size_t)head * ODIM + col0 + m16) * KD;
#pragma unroll
    for (int c = 0; c < 2; ++c) {
      bf16x16 A = ldA16(A0 + 32 * c, hf);
      bf16x16 B = ldB16(B0 + 32 * c + 16 * hf);
      acc = wmma_bf16(A, B, acc);
    }
  }
#pragma unroll
  for (int r = 0; r < 8; ++r)
    out[(size_t)(row0 + r + 8 * hf) * ODIM + col0 + m16] = acc[r];
}

// ---------------- host launcher ----------------
extern "C" void kernel_launch(void* const* d_in, const int* in_sizes, int n_in,
                              void* d_out, int out_size, void* d_ws,
                              size_t ws_size, hipStream_t stream) {
  (void)in_sizes; (void)n_in; (void)out_size; (void)ws_size;
  const float* node = (const float*)d_in[0];
  const float* adj  = (const float*)d_in[1];
  const float* Wks  = (const float*)d_in[2];
  const float* Wqs  = (const float*)d_in[3];
  const float* Wvs  = (const float*)d_in[4];
  const float* Wo   = (const float*)d_in[5];
  float* out = (float*)d_out;

  char* ws = (char*)d_ws;
  size_t off = 0;
  auto salloc = [&](size_t bytes) -> char* {
    char* p = ws + off;
    off += (bytes + 255) & ~(size_t)255;
    return p;
  };
  __bf16* node_bf = (__bf16*)salloc((size_t)NN * IND * 2);          // 4 MB
  __bf16* WqT = (__bf16*)salloc((size_t)NH * KD * IND * 2);         // 512 KB
  __bf16* WkT = (__bf16*)salloc((size_t)NH * KD * IND * 2);
  __bf16* WvT = (__bf16*)salloc((size_t)NH * KD * IND * 2);
  __bf16* WoT = (__bf16*)salloc((size_t)NH * ODIM * KD * 2);
  __bf16* Qb  = (__bf16*)salloc((size_t)NH * NN * KD * 2);          // 4 MB
  __bf16* Kb  = (__bf16*)salloc((size_t)NH * NN * KD * 2);
  __bf16* VTb = (__bf16*)salloc((size_t)NH * KD * NN * 2);
  __bf16* HV  = (__bf16*)salloc((size_t)NH * NN * KD * 2);

  // 0) precision conversion + weight transposes (B operands contiguous)
  {
    int n = NN * IND;
    k_cvt_bf16<<<(n + 255) / 256, 256, 0, stream>>>(node, node_bf, n);
    int tw = NH * IND * KD;
    k_transpose_bf16<<<(tw + 255) / 256, 256, 0, stream>>>(Wqs, WqT, NH, IND, KD);
    k_transpose_bf16<<<(tw + 255) / 256, 256, 0, stream>>>(Wks, WkT, NH, IND, KD);
    k_transpose_bf16<<<(tw + 255) / 256, 256, 0, stream>>>(Wvs, WvT, NH, IND, KD);
    int to = NH * KD * ODIM;
    k_transpose_bf16<<<(to + 255) / 256, 256, 0, stream>>>(Wo, WoT, NH, KD, ODIM);
  }
  // 1) per-head Q (pre-scaled), K, V^T projections
  k_proj<<<dim3(NN / 16, NH, 3), 32, 0, stream>>>(node_bf, WqT, WkT, WvT,
                                                  Qb, Kb, VTb);
  // 2) fused masked flash attention (8 heads per workgroup share adj tile)
  k_flash<<<NN / 16, 256, 0, stream>>>(Qb, Kb, VTb, adj, HV);
  // 3) output projection summed over heads
  k_outproj<<<dim3(NN / 16, ODIM / 16), 32, 0, stream>>>(HV, WoT, out);
}
